// GPTModel_8778913153363
// MI455X (gfx1250) — matrix operand order repaired and checked by reading, not compiled, not run on hardware
//
#include <hip/hip_runtime.h>

typedef __attribute__((ext_vector_type(16))) __bf16 v16bf;
typedef __attribute__((ext_vector_type(8)))  __bf16 v8bf;
typedef __attribute__((ext_vector_type(8)))  float  v8f;

#define D_   1024
#define H_   16
#define DK_  64
#define F_   4096
#define S_   1024
#define B_   2
#define M_   (B_*S_)
#define V_   32000
#define L_   4
#define LN_EPS 1e-5f

// ISA 7.12.2: 16-bit A-matrix 16x32 element map. lane group g = lane>>4.
__device__ __forceinline__ int a_kmap(int e, int g) {
  return (e < 8) ? (g * 8 + e) : (16 + g * 8 + (e - 8));
}

// Raw wave-relative LDS byte offset of a generic pointer to __shared__ data.
// ISA 10.2 aperture rule: the LDS address is the low 32 bits of the flat address.
__device__ __forceinline__ unsigned lds_offset(const void* p) {
  return (unsigned)(unsigned long long)p;
}

// CDNA5 async copy: LDS[lds_off + lane's slot] = MEM[gaddr] (16B per lane),
// tracked by ASYNCcnt (ISA 15.18.3 op 98).
__device__ __forceinline__ void async_load_b128(unsigned lds_off, const void* g) {
  asm volatile("global_load_async_to_lds_b128 %0, %1, off"
               :: "v"(lds_off), "v"(g) : "memory");
}

// -------------------------------------------------------------------------
// Embedding: x[b,s,:] = tok_emb[ids[b,s],:] + pos_emb[s,:]   (fp32 residual)
// -------------------------------------------------------------------------
__global__ void embed_kernel(const int* __restrict__ ids,
                             const float* __restrict__ tok,
                             const float* __restrict__ pos,
                             float* __restrict__ x) {
  int idx = blockIdx.x * blockDim.x + threadIdx.x;   // over M_*D_
  int row = idx / D_, d = idx - row * D_;
  int s = row % S_;
  int id = ids[row];
  x[idx] = tok[(size_t)id * D_ + d] + pos[(size_t)s * D_ + d];
}

// -------------------------------------------------------------------------
// Pack fp32 B matrix into bf16 WMMA B-fragment order:
//   out[tile*512 + lane*16 + e] = B[k = kt*32 + (lane>>4)*16 + e,
//                                   n = nt*16 + (lane&15)]
// TRANSB=1: source stored [N,K] row-major (tied LM head).
// -------------------------------------------------------------------------
template <bool TRANSB>
__global__ void pack_b_kernel(const float* __restrict__ W,
                              __bf16* __restrict__ out, int K, int N) {
  int idx = blockIdx.x * blockDim.x + threadIdx.x;   // over K*N
  int tile = idx >> 9;
  int r = idx & 511;
  int lane = r >> 4, e = r & 15;
  int tilesN = N >> 4;
  int kt = tile / tilesN, nt = tile - kt * tilesN;
  int n = nt * 16 + (lane & 15);
  int k = kt * 32 + (lane >> 4) * 16 + e;
  float v = TRANSB ? W[(size_t)n * K + k] : W[(size_t)k * N + n];
  out[idx] = (__bf16)v;
}

// -------------------------------------------------------------------------
// LayerNorm (fp32 in, bf16 out): one wave per row of D_=1024
// -------------------------------------------------------------------------
__global__ void ln_kernel(const float* __restrict__ x,
                          const float* __restrict__ gg,
                          const float* __restrict__ bb,
                          __bf16* __restrict__ y, int rows) {
  int lane = threadIdx.x & 31;
  int w = threadIdx.x >> 5;
  int row = blockIdx.x * 8 + w;
  if (row >= rows) return;
  const float* xr = x + (size_t)row * D_;
  float v[32];
  float s = 0.f;
#pragma unroll
  for (int i = 0; i < 32; i++) { v[i] = xr[lane + i * 32]; s += v[i]; }
#pragma unroll
  for (int m = 16; m >= 1; m >>= 1) s += __shfl_xor(s, m, 32);
  float mean = s * (1.f / D_);
  float vs = 0.f;
#pragma unroll
  for (int i = 0; i < 32; i++) { float t = v[i] - mean; vs += t * t; }
#pragma unroll
  for (int m = 16; m >= 1; m >>= 1) vs += __shfl_xor(vs, m, 32);
  float rstd = rsqrtf(vs * (1.f / D_) + LN_EPS);
  __bf16* yr = y + (size_t)row * D_;
#pragma unroll
  for (int i = 0; i < 32; i++) {
    int c = lane + i * 32;
    yr[c] = (__bf16)((v[i] - mean) * rstd * gg[c] + bb[c]);
  }
}

// -------------------------------------------------------------------------
// bf16 WMMA GEMM with async-staged shared B:
//   Block = 8 waves covering a 128(M) x 64(N) macro-tile; all waves share the
//   same 4KB B k-slab, double-buffered in LDS via GLOBAL_LOAD_ASYNC_TO_LDS_B128
//   (each wave copies one 512B slice per slab).  A: bf16 [M,K] row-major,
//   fragment-shaped 16B loads.  Bsw: bf16 pre-packed fragment order.
//   C = act(A@B + bias) (+ fp32 resid); OUTBF selects bf16/fp32 store.
// Requires M%128==0, N%64==0, K%32==0; grid sized exactly.
// -------------------------------------------------------------------------
template <bool RELU, bool OUTBF>
__global__ void gemm_wmma_kernel(const __bf16* __restrict__ A,
                                 const __bf16* __restrict__ Bsw,
                                 const float* __restrict__ bias,
                                 const float* __restrict__ resid,
                                 void* __restrict__ Cout,
                                 int Mdim, int N, int K) {
  __shared__ __bf16 ldsb[2 * 2048];   // 2 x (4 tiles x 512 bf16) = 8KB
  int lane = threadIdx.x & 31;
  int w = threadIdx.x >> 5;
  int tilesN64 = N >> 6;
  int bm = blockIdx.x / tilesN64;
  int bn = blockIdx.x - bm * tilesN64;
  int m = lane & 15, g = lane >> 4;
  int row = bm * 128 + w * 16 + m;
  int tilesN16 = N >> 4;
  unsigned lbase = lds_offset(ldsb);
  int slice = w * 512 + lane * 16;         // bytes within a 4KB slab

  v8f acc[4];
#pragma unroll
  for (int t = 0; t < 4; t++)
#pragma unroll
    for (int r = 0; r < 8; r++) acc[t][r] = 0.f;

  const __bf16* Arow = A + (size_t)row * K;
  int nk = K >> 5;
  // prologue: stage slab 0
  {
    const char* gsrc = (const char*)(Bsw + ((size_t)0 * tilesN16 + (bn << 2)) * 512);
    async_load_b128(lbase + slice, gsrc + slice);
  }
  for (int i = 0; i < nk; i++) {
    int buf = i & 1;
    if (i + 1 < nk) {
      const char* gsrc =
          (const char*)(Bsw + ((size_t)(i + 1) * tilesN16 + (bn << 2)) * 512);
      async_load_b128(lbase + ((buf ^ 1) * 4096) + slice, gsrc + slice);
      asm volatile("s_wait_asynccnt 0x1" ::: "memory");  // slab i landed
    } else {
      asm volatile("s_wait_asynccnt 0x0" ::: "memory");
    }
    __syncthreads();

    // A fragment: two contiguous 16B runs of this wave's 16 rows
    const __bf16* ar = Arow + i * 32;
    v8bf r0 = *(const v8bf*)(ar + g * 16);
    v8bf r1 = *(const v8bf*)(ar + g * 16 + 8);
    v16bf af = __builtin_shufflevector(r0, r1, 0, 1, 2, 3, 4, 5, 6, 7,
                                       8, 9, 10, 11, 12, 13, 14, 15);
    const __bf16* lb = ldsb + buf * 2048 + lane * 16;
#pragma unroll
    for (int t = 0; t < 4; t++) {
      v16bf bf_ = *(const v16bf*)(lb + t * 512);
      acc[t] = __builtin_amdgcn_wmma_f32_16x16x32_bf16(
          false, af, false, bf_, (short)0, acc[t], false, false);
    }
    __syncthreads();   // all waves done reading slab i before it is re-staged
  }

#pragma unroll
  for (int t = 0; t < 4; t++) {
    int col = (bn << 6) + t * 16 + m;
    float bs = bias ? bias[col] : 0.f;
#pragma unroll
    for (int r = 0; r < 8; r++) {
      int orow = bm * 128 + w * 16 + g * 8 + r;
      float vv = acc[t][r] + bs;
      if (RELU) vv = fmaxf(vv, 0.f);
      size_t oi = (size_t)orow * N + col;
      if (resid) vv += resid[oi];
      if (OUTBF) ((__bf16*)Cout)[oi] = (__bf16)vv;
      else       ((float*)Cout)[oi] = vv;
    }
  }
}

// -------------------------------------------------------------------------
// Flash attention, bf16 q/k/v/ctx [B,S,D] (head h = cols h*DK..h*DK+63).
// One wave per (b, h, 16-query tile); WMMA for QK^T and PV; online softmax
// via 16-lane shfl reductions; P re-striped C-layout -> A-layout via LDS.
// -------------------------------------------------------------------------
__global__ void attn_kernel(const __bf16* __restrict__ q,
                            const __bf16* __restrict__ k,
                            const __bf16* __restrict__ v,
                            __bf16* __restrict__ ctx) {
  __shared__ __bf16 pbuf[8][16][32];
  int lane = threadIdx.x & 31;
  int wid = threadIdx.x >> 5;
  int wvg = blockIdx.x * 8 + wid;     // B_*H_*(S_/16) waves total
  int qb = wvg & 63;                  // S_/16 = 64
  int h = (wvg >> 6) & 15;
  int bb = wvg >> 10;
  int m = lane & 15, g = lane >> 4;

  const __bf16* qbase = q + ((size_t)(bb * S_ + qb * 16 + m)) * D_ + h * DK_;
  v16bf aq[2];
#pragma unroll
  for (int kh = 0; kh < 2; kh++) {
    v8bf r0 = *(const v8bf*)(qbase + kh * 32 + g * 16);
    v8bf r1 = *(const v8bf*)(qbase + kh * 32 + g * 16 + 8);
    aq[kh] = __builtin_shufflevector(r0, r1, 0, 1, 2, 3, 4, 5, 6, 7,
                                     8, 9, 10, 11, 12, 13, 14, 15);
  }

  float mi[8], li[8];
  v8f o[4];
#pragma unroll
  for (int r = 0; r < 8; r++) { mi[r] = -1e30f; li[r] = 0.f; }
#pragma unroll
  for (int t = 0; t < 4; t++)
#pragma unroll
    for (int r = 0; r < 8; r++) o[t][r] = 0.f;

  int nkb = (qb >> 1) + 1;
  for (int kb = 0; kb < nkb; kb++) {
    int kv0 = kb * 32;
    v8f sc[2];
#pragma unroll
    for (int t = 0; t < 2; t++) {
      v8f s;
#pragma unroll
      for (int r = 0; r < 8; r++) s[r] = 0.f;
      const __bf16* kbase =
          k + ((size_t)(bb * S_ + kv0 + t * 16 + m)) * D_ + h * DK_;
#pragma unroll
      for (int kh = 0; kh < 2; kh++) {
        v16bf bk = *(const v16bf*)(kbase + kh * 32 + g * 16);
        s = __builtin_amdgcn_wmma_f32_16x16x32_bf16(
            false, aq[kh], false, bk, (short)0, s, false, false);
      }
      sc[t] = s;
    }
#pragma unroll
    for (int r = 0; r < 8; r++) {
      int qr = qb * 16 + g * 8 + r;
#pragma unroll
      for (int t = 0; t < 2; t++) {
        int kc = kv0 + t * 16 + m;
        float sv = sc[t][r] * 0.125f;          // 1/sqrt(DK)
        sc[t][r] = (kc > qr) ? -1e30f : sv;
      }
      float tmax = fmaxf(sc[0][r], sc[1][r]);
#pragma unroll
      for (int msk = 8; msk >= 1; msk >>= 1)
        tmax = fmaxf(tmax, __shfl_xor(tmax, msk, 32));
      float mnew = fmaxf(mi[r], tmax);
      float p0 = __expf(sc[0][r] - mnew);
      float p1 = __expf(sc[1][r] - mnew);
      sc[0][r] = p0; sc[1][r] = p1;
      float ssum = p0 + p1;
#pragma unroll
      for (int msk = 8; msk >= 1; msk >>= 1) ssum += __shfl_xor(ssum, msk, 32);
      float alpha = __expf(mi[r] - mnew);
      li[r] = li[r] * alpha + ssum;
      mi[r] = mnew;
#pragma unroll
      for (int t = 0; t < 4; t++) o[t][r] *= alpha;
    }
#pragma unroll
    for (int t = 0; t < 2; t++)
#pragma unroll
      for (int r = 0; r < 8; r++)
        pbuf[wid][g * 8 + r][t * 16 + m] = (__bf16)sc[t][r];
    asm volatile("s_wait_dscnt 0x0" ::: "memory");
    v16bf ap;
#pragma unroll
    for (int e = 0; e < 16; e++) ap[e] = pbuf[wid][m][g * 16 + e];
#pragma unroll
    for (int td = 0; td < 4; td++) {
      const __bf16* vb = v + ((size_t)(bb * S_ + kv0 + g * 16)) * D_ +
                         h * DK_ + td * 16 + m;
      v16bf bv;
#pragma unroll
      for (int e = 0; e < 16; e++) bv[e] = vb[(size_t)e * D_];
      o[td] = __builtin_amdgcn_wmma_f32_16x16x32_bf16(
          false, ap, false, bv, (short)0, o[td], false, false);
    }
  }
#pragma unroll
  for (int td = 0; td < 4; td++)
#pragma unroll
    for (int r = 0; r < 8; r++) {
      int srow = qb * 16 + g * 8 + r;
      ctx[((size_t)(bb * S_ + srow)) * D_ + h * DK_ + td * 16 + m] =
          (__bf16)(o[td][r] / li[r]);
    }
}

// -------------------------------------------------------------------------
// Host-side dispatch
// -------------------------------------------------------------------------
static void run_gemm(const __bf16* A, const __bf16* Bsw, const float* bias,
                     const float* resid, void* C, int M, int N, int K,
                     bool relu, bool outbf, hipStream_t st) {
  int blocks = (M / 128) * (N / 64);
  if (outbf) {
    if (relu)
      gemm_wmma_kernel<true, true><<<blocks, 256, 0, st>>>(A, Bsw, bias, resid, C, M, N, K);
    else
      gemm_wmma_kernel<false, true><<<blocks, 256, 0, st>>>(A, Bsw, bias, resid, C, M, N, K);
  } else {
    if (relu)
      gemm_wmma_kernel<true, false><<<blocks, 256, 0, st>>>(A, Bsw, bias, resid, C, M, N, K);
    else
      gemm_wmma_kernel<false, false><<<blocks, 256, 0, st>>>(A, Bsw, bias, resid, C, M, N, K);
  }
}

static void run_pack(const float* W, __bf16* out, int K, int N, bool transb,
                     hipStream_t st) {
  int blocks = (K * N) / 256;
  if (transb) pack_b_kernel<true><<<blocks, 256, 0, st>>>(W, out, K, N);
  else        pack_b_kernel<false><<<blocks, 256, 0, st>>>(W, out, K, N);
}

extern "C" void kernel_launch(void* const* d_in, const int* in_sizes, int n_in,
                              void* d_out, int out_size, void* d_ws, size_t ws_size,
                              hipStream_t stream) {
  (void)in_sizes; (void)n_in; (void)out_size; (void)ws_size;
  const int*   ids  = (const int*)d_in[0];
  const float* tok  = (const float*)d_in[1];
  const float* pos  = (const float*)d_in[2];
  const float* Wq = (const float*)d_in[3];  const float* bq = (const float*)d_in[4];
  const float* Wk = (const float*)d_in[5];  const float* bk = (const float*)d_in[6];
  const float* Wv = (const float*)d_in[7];  const float* bv = (const float*)d_in[8];
  const float* Wo = (const float*)d_in[9];  const float* bo = (const float*)d_in[10];
  const float* ln1g = (const float*)d_in[11]; const float* ln1b = (const float*)d_in[12];
  const float* ln2g = (const float*)d_in[13]; const float* ln2b = (const float*)d_in[14];
  const float* W1 = (const float*)d_in[15]; const float* b1 = (const float*)d_in[16];
  const float* W2 = (const float*)d_in[17]; const float* b2 = (const float*)d_in[18];
  const float* lnfg = (const float*)d_in[19]; const float* lnfb = (const float*)d_in[20];
  float* out = (float*)d_out;

  // ---- workspace carve ----
  char* p = (char*)d_ws;
  size_t MD = (size_t)M_ * D_;
  size_t MF = (size_t)M_ * F_;
  float*  x   = (float*)p;   p += MD * 4;
  __bf16* nx  = (__bf16*)p;  p += MD * 2;
  __bf16* qb_ = (__bf16*)p;  p += MD * 2;
  __bf16* kb_ = (__bf16*)p;  p += MD * 2;
  __bf16* vb_ = (__bf16*)p;  p += MD * 2;
  __bf16* cb_ = (__bf16*)p;  p += MD * 2;
  __bf16* hb_ = (__bf16*)p;  p += MF * 2;
  __bf16* swq = (__bf16*)p;  p += (size_t)D_ * D_ * 2;
  __bf16* swk = (__bf16*)p;  p += (size_t)D_ * D_ * 2;
  __bf16* swv = (__bf16*)p;  p += (size_t)D_ * D_ * 2;
  __bf16* swo = (__bf16*)p;  p += (size_t)D_ * D_ * 2;
  __bf16* sw1 = (__bf16*)p;  p += (size_t)D_ * F_ * 2;
  __bf16* sw2 = (__bf16*)p;  p += (size_t)F_ * D_ * 2;
  __bf16* stok = (__bf16*)p; p += (size_t)D_ * V_ * 2;

  embed_kernel<<<(M_ * D_) / 256, 256, 0, stream>>>(ids, tok, pos, x);
  run_pack(tok, stok, D_, V_, true, stream);

  for (int l = 0; l < L_; l++) {
    run_pack(Wq + (size_t)l * D_ * D_, swq, D_, D_, false, stream);
    run_pack(Wk + (size_t)l * D_ * D_, swk, D_, D_, false, stream);
    run_pack(Wv + (size_t)l * D_ * D_, swv, D_, D_, false, stream);
    run_pack(Wo + (size_t)l * D_ * D_, swo, D_, D_, false, stream);
    run_pack(W1 + (size_t)l * D_ * F_, sw1, D_, F_, false, stream);
    run_pack(W2 + (size_t)l * F_ * D_, sw2, F_, D_, false, stream);

    ln_kernel<<<M_ / 8, 256, 0, stream>>>(x, ln1g + l * D_, ln1b + l * D_, nx, M_);
    run_gemm(nx, swq, bq + l * D_, nullptr, qb_, M_, D_, D_, false, true, stream);
    run_gemm(nx, swk, bk + l * D_, nullptr, kb_, M_, D_, D_, false, true, stream);
    run_gemm(nx, swv, bv + l * D_, nullptr, vb_, M_, D_, D_, false, true, stream);
    attn_kernel<<<(B_ * H_ * (S_ / 16)) / 8, 256, 0, stream>>>(qb_, kb_, vb_, cb_);
    run_gemm(cb_, swo, bo + l * D_, x, x, M_, D_, D_, false, false, stream);
    ln_kernel<<<M_ / 8, 256, 0, stream>>>(x, ln2g + l * D_, ln2b + l * D_, nx, M_);
    run_gemm(nx, sw1, b1 + l * F_, nullptr, hb_, M_, F_, D_, true, true, stream);
    run_gemm(hb_, sw2, b2 + l * D_, x, x, M_, D_, F_, false, false, stream);
  }

  ln_kernel<<<M_ / 8, 256, 0, stream>>>(x, lnfg, lnfb, nx, M_);
  run_gemm(nx, stok, nullptr, nullptr, out, M_, V_, D_, false, false, stream);
}
